// LSTM_54571854463106
// MI455X (gfx1250) — compile-verified, matrix operand order checked by reading, not run on hardware
//
#include <hip/hip_runtime.h>
#include <hip/hip_bf16.h>
#include <math.h>
#include <stdint.h>

// ---------------------------------------------------------------------------
// CDNA5 (gfx1250) WMMA bf16 LSTM-variant, fragment-swizzled operands.
//   All GEMM operands are pre-repacked into the ISA 7.12.2 bf16 fragment
//   layout so the inner loop is just b128 loads + v_wmma (no VALU packing).
//   Precompute waves block 4 column tiles to reuse A fragments 4-8x.
// ---------------------------------------------------------------------------

typedef __attribute__((ext_vector_type(16))) __bf16 v16bf;
typedef __attribute__((ext_vector_type(8)))  float  v8f;

#define T_LEN  256
#define BATCH  64
#define K_DIM  512    // both IN_DIM and MEM_DIM
#define MEM    512
#define G4     2048   // 4*MEM

// frag layout: [tile][kstep(16)][lane(32)][8 dwords]  (4096 dwords per tile)
#define TILE_DW 4096

union FragU { uint32_t u[8]; uint4 q[2]; v16bf v; };

__device__ __forceinline__ uint32_t pack_bf16(float lo, float hi) {
  uint32_t a = __float_as_uint(lo);
  uint32_t b = __float_as_uint(hi);
  a = (a + 0x7FFFu + ((a >> 16) & 1u)) >> 16;
  b = (b + 0x7FFFu + ((b >> 16) & 1u)) >> 16;
  return (a & 0xFFFFu) | (b << 16);
}

__device__ __forceinline__ uint16_t f2bf(float x) {
  uint32_t u = __float_as_uint(x);
  return (uint16_t)((u + 0x7FFFu + ((u >> 16) & 1u)) >> 16);
}

__device__ __forceinline__ v16bf ld_frag(const uint32_t* __restrict__ p) {
  FragU f;
  f.q[0] = *(const uint4*)(p);
  f.q[1] = *(const uint4*)(p + 4);
  return f.v;
}

__device__ __forceinline__ v8f wmma_bf16(v16bf a, v16bf b, v8f c) {
  return __builtin_amdgcn_wmma_f32_16x16x32_bf16(false, a, false, b,
                                                 (short)0, c, false, false);
}

__device__ __forceinline__ float sigmoidf_(float x) {
  return 1.0f / (1.0f + __expf(-x));
}

// ---------------------------------------------------------------------------
// Repack a row-major f32 matrix [N x 512] into bf16 WMMA fragment order.
// dword i decomposes as: v=i[2:0], lane=i[7:3], kstep=i[11:8], tile=i[31:12].
// A-matrix: row = tile*16 + lane%16;  B-matrix identical since B[k][n]=W[n][k].
// ---------------------------------------------------------------------------
__global__ void __launch_bounds__(256)
k_swizzle(const float* __restrict__ src, uint32_t* __restrict__ dst, int n_dw) {
  for (int i = blockIdx.x * blockDim.x + threadIdx.x; i < n_dw;
       i += gridDim.x * blockDim.x) {
    const int v    = i & 7;
    const int lane = (i >> 3) & 31;
    const int ks   = (i >> 8) & 15;
    const int tile = i >> 12;
    const int half = lane >> 4;
    const int k    = ks * 32 + 8 * half + (v < 4 ? 2 * v : 16 + 2 * (v - 4));
    const int row  = tile * 16 + (lane & 15);
    const float2 x = *(const float2*)(src + (size_t)row * K_DIM + k);
    dst[i] = pack_bf16(x.x, x.y);
  }
}

// h0 f32 -> plain row-major bf16 (slot 0 of the per-timestep h ring)
__global__ void __launch_bounds__(256)
k_cvt_h0(const float* __restrict__ src, uint16_t* __restrict__ dst, int n) {
  const int i = blockIdx.x * blockDim.x + threadIdx.x;
  if (i < n) dst[i] = f2bf(src[i]);
}

// ---------------------------------------------------------------------------
// Phase 1a: iou_x[T*B, 2048] = X@Wioux^T + S@Wious^T + b_ioux + b_ious
// One wave = 1 row tile x 4 col tiles; A fragments reused across 8 WMMAs.
// ---------------------------------------------------------------------------
__global__ void __launch_bounds__(256)
k_precompute_iou(const uint32_t* __restrict__ Xf, const uint32_t* __restrict__ Sf,
                 const uint32_t* __restrict__ Wxf, const uint32_t* __restrict__ Wsf,
                 const float* __restrict__ bx, const float* __restrict__ bs,
                 float* __restrict__ iou_x) {
  const int lane = threadIdx.x & 31;
  const int gw   = blockIdx.x * 8 + (threadIdx.x >> 5);
  const int rt   = gw >> 5;           // 1024 row tiles
  const int ct0  = (gw & 31) * 4;     // 32 groups of 4 col tiles
  const uint32_t* ax = Xf  + (size_t)rt * TILE_DW + lane * 8;
  const uint32_t* as = Sf  + (size_t)rt * TILE_DW + lane * 8;
  const uint32_t* bw = Wxf + (size_t)ct0 * TILE_DW + lane * 8;
  const uint32_t* bv = Wsf + (size_t)ct0 * TILE_DW + lane * 8;
  v8f accs[4] = {v8f{}, v8f{}, v8f{}, v8f{}};
#pragma unroll 2
  for (int kk = 0; kk < 16; ++kk) {
    __builtin_prefetch(ax + 256, 0, 3);
    __builtin_prefetch(as + 256, 0, 3);
    const v16bf aX = ld_frag(ax);
    const v16bf aS = ld_frag(as);
#pragma unroll
    for (int c = 0; c < 4; ++c) {
      accs[c] = wmma_bf16(aX, ld_frag(bw + c * TILE_DW), accs[c]);
      accs[c] = wmma_bf16(aS, ld_frag(bv + c * TILE_DW), accs[c]);
    }
    ax += 256; as += 256; bw += 256; bv += 256;
  }
  const int half = lane >> 4, ln = lane & 15;
  const int row0 = rt * 16;
#pragma unroll
  for (int c = 0; c < 4; ++c) {
    const int col0 = (ct0 + c) * 16;
    const float bias = bx[col0 + ln] + bs[col0 + ln];
#pragma unroll
    for (int v = 0; v < 8; ++v)
      iou_x[(size_t)(row0 + v + 8 * half) * G4 + col0 + ln] = accs[c][v] + bias;
  }
}

// ---------------------------------------------------------------------------
// Phase 1b: ctx = X@Wfx^T + b_fx ; wcs = tanh(S@Wwc^T + b_wc)
// One wave = 1 row tile x 4 col tiles; col groups 0..7 -> ctx, 8..15 -> wcs.
// ---------------------------------------------------------------------------
__global__ void __launch_bounds__(256)
k_precompute_ct_wc(const uint32_t* __restrict__ Xf, const uint32_t* __restrict__ Sf,
                   const uint32_t* __restrict__ Wfxf, const uint32_t* __restrict__ Wwcf,
                   const float* __restrict__ bfx, const float* __restrict__ bwc,
                   float* __restrict__ ctx, float* __restrict__ wcs) {
  const int lane = threadIdx.x & 31;
  const int gw   = blockIdx.x * 8 + (threadIdx.x >> 5);
  const int rt   = gw >> 4;           // 1024 row tiles
  const int cg   = gw & 15;           // 8 ctx groups + 8 wcs groups
  const bool is_wc = (cg >= 8);
  const int ct0  = (is_wc ? (cg - 8) : cg) * 4;
  const uint32_t* ap = (is_wc ? Sf : Xf)     + (size_t)rt * TILE_DW + lane * 8;
  const uint32_t* bp = (is_wc ? Wwcf : Wfxf) + (size_t)ct0 * TILE_DW + lane * 8;
  v8f accs[4] = {v8f{}, v8f{}, v8f{}, v8f{}};
#pragma unroll 2
  for (int kk = 0; kk < 16; ++kk) {
    __builtin_prefetch(ap + 256, 0, 3);
    const v16bf a = ld_frag(ap);
#pragma unroll
    for (int c = 0; c < 4; ++c)
      accs[c] = wmma_bf16(a, ld_frag(bp + c * TILE_DW), accs[c]);
    ap += 256; bp += 256;
  }
  const int half = lane >> 4, ln = lane & 15;
  const int row0 = rt * 16;
  if (is_wc) {
#pragma unroll
    for (int c = 0; c < 4; ++c) {
      const int col0 = (ct0 + c) * 16;
      const float bias = bwc[col0 + ln];
#pragma unroll
      for (int v = 0; v < 8; ++v)
        wcs[(size_t)(row0 + v + 8 * half) * MEM + col0 + ln] = tanhf(accs[c][v] + bias);
    }
  } else {
#pragma unroll
    for (int c = 0; c < 4; ++c) {
      const int col0 = (ct0 + c) * 16;
      const float bias = bfx[col0 + ln];
#pragma unroll
      for (int v = 0; v < 8; ++v)
        ctx[(size_t)(row0 + v + 8 * half) * MEM + col0 + ln] = accs[c][v] + bias;
    }
  }
}

// ---------------------------------------------------------------------------
// Phase 2: one fused timestep. Block = 16 cells (5 gate tiles), wave = 16
// batch rows. A = h_{t-1} in plain bf16 (dword loads, immediate k offsets);
// B = swizzled W_iouh/W_fh with compile-time gate offsets (fit vmem imm24).
// Writes: out f32, h bf16 slot t+1, c state.
// ---------------------------------------------------------------------------
__global__ void __launch_bounds__(128)
k_lstm_step(const uint16_t* __restrict__ h_prev_bf, uint16_t* __restrict__ h_next_bf,
            float* __restrict__ c_state,
            const uint32_t* __restrict__ Wiouhf, const uint32_t* __restrict__ Wfhf,
            const float* __restrict__ b_iouh, const float* __restrict__ b_fh,
            const float* __restrict__ iou_x, const float* __restrict__ ctx,
            const float* __restrict__ wcs, float* __restrict__ out_h, int t) {
  const int lane = threadIdx.x & 31;
  const int half = lane >> 4, ln = lane & 15;
  const int wave = threadIdx.x >> 5;
  const int blk  = blockIdx.x;      // 32 blocks cover MEM cells
  const int j0   = blk * 16;
  const int r0   = wave * 16;       // 4 waves cover BATCH rows
  // A: lane base into h bf16 (row-major); dword offsets are compile-time.
  const uint32_t* hp = (const uint32_t*)(h_prev_bf + (size_t)(r0 + ln) * MEM) + half * 4;
  // B: gate g lives at W_iouh rows [g*512, g*512+512) -> tile g*32 + blk.
  const uint32_t* bW = Wiouhf + (size_t)blk * TILE_DW + lane * 8;  // + g*32*TILE_DW
  const uint32_t* bT = Wfhf   + (size_t)blk * TILE_DW + lane * 8;
  v8f aF = {}, aI = {}, aO = {}, aC = {}, aT = {};
#pragma unroll 2
  for (int kk = 0; kk < 16; ++kk) {
    FragU a;
    const int d = kk * 16;
#pragma unroll
    for (int v = 0; v < 4; ++v) a.u[v] = hp[d + v];
#pragma unroll
    for (int v = 4; v < 8; ++v) a.u[v] = hp[d + 4 + v];
    aF = wmma_bf16(a.v, ld_frag(bW + 0 * 32 * TILE_DW), aF);
    aI = wmma_bf16(a.v, ld_frag(bW + 1 * 32 * TILE_DW), aI);
    aO = wmma_bf16(a.v, ld_frag(bW + 2 * 32 * TILE_DW), aO);
    aC = wmma_bf16(a.v, ld_frag(bW + 3 * 32 * TILE_DW), aC);
    aT = wmma_bf16(a.v, ld_frag(bT), aT);
    bW += 256; bT += 256;
  }
  const int j  = j0 + ln;
  const float bF = b_iouh[0 * MEM + j];
  const float bI = b_iouh[1 * MEM + j];
  const float bO = b_iouh[2 * MEM + j];
  const float bC = b_iouh[3 * MEM + j];
  const float bTt = b_fh[j];
#pragma unroll
  for (int v = 0; v < 8; ++v) {
    const int br = r0 + v + 8 * half;               // batch row
    const size_t rg = (size_t)t * BATCH + br;       // row in [T*B] space
    const float f   = sigmoidf_(aF[v] + iou_x[rg * G4 + 0 * MEM + j] + bF);
    const float i   = sigmoidf_(aI[v] + iou_x[rg * G4 + 1 * MEM + j] + bI);
    const float o   = sigmoidf_(aO[v] + iou_x[rg * G4 + 2 * MEM + j] + bO);
    const float oc  = sigmoidf_(aC[v] + iou_x[rg * G4 + 3 * MEM + j] + bC);
    const float ctl = tanhf(aT[v] + ctx[rg * MEM + j] + bTt);
    const float c_old = c_state[(size_t)br * MEM + j];
    const float c_new = i * ctl + f * c_old;
    c_state[(size_t)br * MEM + j] = c_new;
    const float h_new = o * tanhf(c_new) + oc * wcs[rg * MEM + j];
    out_h[rg * MEM + j] = h_new;
    h_next_bf[(size_t)br * MEM + j] = f2bf(h_new);
  }
}

// ---------------------------------------------------------------------------
extern "C" void kernel_launch(void* const* d_in, const int* in_sizes, int n_in,
                              void* d_out, int out_size, void* d_ws, size_t ws_size,
                              hipStream_t stream) {
  const float* inputs  = (const float*)d_in[0];
  const float* emb_s   = (const float*)d_in[1];
  const float* c0      = (const float*)d_in[2];
  const float* h0      = (const float*)d_in[3];
  const float* W_ioux  = (const float*)d_in[4];
  const float* b_ioux  = (const float*)d_in[5];
  const float* W_iouh  = (const float*)d_in[6];
  const float* b_iouh  = (const float*)d_in[7];
  const float* W_ious  = (const float*)d_in[8];
  const float* b_ious  = (const float*)d_in[9];
  const float* W_fx    = (const float*)d_in[10];
  const float* b_fx    = (const float*)d_in[11];
  const float* W_fh    = (const float*)d_in[12];
  const float* b_fh    = (const float*)d_in[13];
  const float* W_wc    = (const float*)d_in[14];
  const float* b_wc    = (const float*)d_in[15];
  float* out = (float*)d_out;

  const size_t R = (size_t)T_LEN * BATCH;     // 16384 rows
  // Workspace carve-out
  char* ws = (char*)d_ws;
  float*    iou_x  = (float*)ws;    ws += R * G4 * sizeof(float);           // 134 MB
  float*    ctx    = (float*)ws;    ws += R * MEM * sizeof(float);          // 33.5 MB
  float*    wcs    = (float*)ws;    ws += R * MEM * sizeof(float);          // 33.5 MB
  float*    c_st   = (float*)ws;    ws += (size_t)BATCH * MEM * sizeof(float);
  uint32_t* Xf     = (uint32_t*)ws; ws += R * K_DIM / 2 * sizeof(uint32_t); // 16.8 MB
  uint32_t* Sf     = (uint32_t*)ws; ws += R * K_DIM / 2 * sizeof(uint32_t);
  uint32_t* Wiouxf = (uint32_t*)ws; ws += (size_t)G4 * K_DIM / 2 * sizeof(uint32_t);
  uint32_t* Wiousf = (uint32_t*)ws; ws += (size_t)G4 * K_DIM / 2 * sizeof(uint32_t);
  uint32_t* Wiouhf = (uint32_t*)ws; ws += (size_t)G4 * K_DIM / 2 * sizeof(uint32_t);
  uint32_t* Wfxf   = (uint32_t*)ws; ws += (size_t)MEM * K_DIM / 2 * sizeof(uint32_t);
  uint32_t* Wfhf   = (uint32_t*)ws; ws += (size_t)MEM * K_DIM / 2 * sizeof(uint32_t);
  uint32_t* Wwcf   = (uint32_t*)ws; ws += (size_t)MEM * K_DIM / 2 * sizeof(uint32_t);
  uint16_t* h_bf   = (uint16_t*)ws; ws += (size_t)(T_LEN + 1) * BATCH * MEM * sizeof(uint16_t);

  // Phase 0: fragment-swizzle all GEMM operands (f32 -> bf16, one pass each).
  {
    const int nA = (int)(R * K_DIM / 2);
    k_swizzle<<<(nA + 255) / 256, 256, 0, stream>>>(inputs, Xf, nA);
    k_swizzle<<<(nA + 255) / 256, 256, 0, stream>>>(emb_s,  Sf, nA);
    const int nW = G4 * K_DIM / 2;
    k_swizzle<<<(nW + 255) / 256, 256, 0, stream>>>(W_ioux, Wiouxf, nW);
    k_swizzle<<<(nW + 255) / 256, 256, 0, stream>>>(W_ious, Wiousf, nW);
    k_swizzle<<<(nW + 255) / 256, 256, 0, stream>>>(W_iouh, Wiouhf, nW);
    const int nM = MEM * K_DIM / 2;
    k_swizzle<<<(nM + 255) / 256, 256, 0, stream>>>(W_fx, Wfxf, nM);
    k_swizzle<<<(nM + 255) / 256, 256, 0, stream>>>(W_fh, Wfhf, nM);
    k_swizzle<<<(nM + 255) / 256, 256, 0, stream>>>(W_wc, Wwcf, nM);
    const int nH = BATCH * MEM;
    k_cvt_h0<<<(nH + 255) / 256, 256, 0, stream>>>(h0, h_bf, nH);
  }

  // Phase 1: time-parallel GEMMs (8 waves/block, 4 col tiles per wave).
  k_precompute_iou<<<(int)(R / 16) * 32 / 8, 256, 0, stream>>>(
      Xf, Sf, Wiouxf, Wiousf, b_ioux, b_ious, iou_x);
  k_precompute_ct_wc<<<(int)(R / 16) * 16 / 8, 256, 0, stream>>>(
      Xf, Sf, Wfxf, Wwcf, b_fx, b_wc, ctx, wcs);

  hipMemcpyAsync(c_st, c0, (size_t)BATCH * MEM * sizeof(float),
                 hipMemcpyDeviceToDevice, stream);

  // Phase 2: sequential scan; h kept in per-t bf16 slots (no races).
  for (int t = 0; t < T_LEN; ++t) {
    k_lstm_step<<<MEM / 16, 128, 0, stream>>>(
        h_bf + (size_t)t * BATCH * MEM, h_bf + (size_t)(t + 1) * BATCH * MEM,
        c_st, Wiouhf, Wfhf, b_iouh, b_fh, iou_x, ctx, wcs, out, t);
  }

  // Tail outputs: (c_fin, h_fin) after the h sequence.
  float* tail = out + (size_t)T_LEN * BATCH * MEM;
  hipMemcpyAsync(tail, c_st, (size_t)BATCH * MEM * sizeof(float),
                 hipMemcpyDeviceToDevice, stream);
  hipMemcpyAsync(tail + (size_t)BATCH * MEM,
                 out + (size_t)(T_LEN - 1) * BATCH * MEM,
                 (size_t)BATCH * MEM * sizeof(float),
                 hipMemcpyDeviceToDevice, stream);
}
